// RSABlock_3006477107754
// MI455X (gfx1250) — matrix operand order repaired and checked by reading, not compiled, not run on hardware
//
#include <hip/hip_runtime.h>
#include <hip/hip_bf16.h>
#include <math.h>
#include <stdint.h>

// ---------------------------------------------------------------------------
// RSA block for MI455X (gfx1250, wave32, WMMA + TDM).
// GEMMs: v_wmma_f32_16x16x32_bf16, 16x64 output per wave (A-frag reuse x4).
// Attention kernel stages Q/K via tensor_load_to_lds (TENSORcnt).
// ---------------------------------------------------------------------------

#define D_DIM   512
#define AGENTS  64
#define BATCH   512
#define M_ROWS  (BATCH * AGENTS)        // 32768
#define NT      (D_DIM / 16)            // 32 tile cols
#define MT      (M_ROWS / 16)           // 2048 tile rows
#define KBLK    (D_DIM / 32)            // 16 k-steps of 32

typedef __attribute__((ext_vector_type(16))) __bf16 v16bf;
typedef __attribute__((ext_vector_type(8)))  float  v8f;

__device__ __forceinline__ uint16_t f32_to_bf16(float x) {
    uint32_t u = __builtin_bit_cast(uint32_t, x);
    uint32_t r = (u + 0x7FFFu + ((u >> 16) & 1u)) >> 16;   // round-nearest-even
    return (uint16_t)r;
}

// Pack two f32 -> packed bf16x2 (one v_cvt_pk_bf16_f32 when available).
__device__ __forceinline__ uint32_t pack_bf16(float lo, float hi) {
#if __has_builtin(__builtin_amdgcn_cvt_pk_bf16_f32)
    typedef __attribute__((ext_vector_type(2))) __bf16 v2bf;
    v2bf r = __builtin_amdgcn_cvt_pk_bf16_f32(lo, hi);
    return __builtin_bit_cast(uint32_t, r);
#else
    return (uint32_t)f32_to_bf16(lo) | ((uint32_t)f32_to_bf16(hi) << 16);
#endif
}

// A-matrix 16x32 bf16 fragment (ISA 7.12.2): lane L holds row M=L&15.
__device__ __forceinline__ v16bf load_fragA(const uint16_t* base, int stride, int lane) {
    union { uint32_t u[8]; v16bf v; } f;
    const int r  = lane & 15;
    const int kh = (lane >> 4) << 3;           // 0 or 8
#pragma unroll
    for (int i = 0; i < 8; ++i) {
        int k0 = ((i & 3) * 2) + ((i >> 2) * 16) + kh;
        f.u[i] = *reinterpret_cast<const uint32_t*>(base + (size_t)r * stride + k0);
    }
    return f.v;
}

// B-matrix 32x16 bf16 fragment from transposed storage Bt[n][k].
__device__ __forceinline__ v16bf load_fragB(const uint16_t* baseT, int stride, int lane) {
    union { uint32_t u[8]; v16bf v; } f;
    const int n  = lane & 15;
    const int kh = (lane >> 4) << 3;           // 0 or 8
#pragma unroll
    for (int i = 0; i < 8; ++i) {
        f.u[i] = *reinterpret_cast<const uint32_t*>(baseT + (size_t)n * stride + ((i & 3) * 2) + ((i >> 2) * 16) + kh);
    }
    return f.v;
}

__device__ __forceinline__ v8f wmma_bf16(v16bf a, v16bf b, v8f c) {
    return __builtin_amdgcn_wmma_f32_16x16x32_bf16(false, a, false, b, (short)0, c, false, false);
}

// ---------------------------------------------------------------------------
// TDM: 1-D tensor_load_to_lds of `nelem` bf16 elements (ISA §8 descriptor).
// count=1, type=2(image), data_size=2B, tile_dim0=tensor_dim0=nelem.
// ---------------------------------------------------------------------------
#if __has_builtin(__builtin_amdgcn_tensor_load_to_lds)
__device__ __forceinline__ void tdm_load_bf16_1d(uint32_t lds_off, const uint16_t* g,
                                                 uint32_t nelem) {
    typedef __attribute__((ext_vector_type(4))) unsigned int tdm_v4u;
    typedef __attribute__((ext_vector_type(8))) int          tdm_v8i;
    typedef __attribute__((ext_vector_type(4))) int          tdm_v4i;
    uint64_t ga = (uint64_t)(uintptr_t)g;
    unsigned hi16 = (nelem & 0xFFFFu) << 16;
    tdm_v4u g0 = { 1u,                                  // count=1 (valid)
                   lds_off,                             // lds_addr (bytes)
                   (unsigned)ga,                        // global_addr[31:0]
                   (unsigned)((ga >> 32) & 0x01FFFFFFu) | (2u << 30) }; // [56:32]|type=2
    tdm_v8i g1 = { (int)(1u << 16),   // data_size=1 (2 bytes); wg_mask=0
                   (int)hi16,         // tensor_dim0[15:0] in bits[63:48]
                   (int)(1u << 16),   // tensor_dim1=1 in bits[95:80]
                   (int)hi16,         // tile_dim0 in bits[127:112]
                   1,                 // tile_dim1=1
                   (int)nelem,        // tensor_dim0_stride[31:0]
                   (int)hi16,         // tensor_dim1_stride[15:0] in bits[223:208]
                   0 };
    tdm_v4i z4 = { 0, 0, 0, 0 };
#if __has_include(<hip/amd_detail/amd_gfx1250_TDM.h>)
    tdm_v8i z8 = { 0, 0, 0, 0, 0, 0, 0, 0 };
    __builtin_amdgcn_tensor_load_to_lds(g0, g1, z4, z4, z8, 0);   // clang-23 form
#else
    __builtin_amdgcn_tensor_load_to_lds(g0, g1, z4, z4, 0);       // ROCm 7.2 form
#endif
}
#endif

// ---------------------------------------------------------------------------
// Kernel 1: transpose + convert one 512x512 f32 weight to bf16 [N][K].
// ---------------------------------------------------------------------------
__global__ void wt_convert(const float* __restrict__ W, uint16_t* __restrict__ WT) {
    int idx = blockIdx.x * blockDim.x + threadIdx.x;
    int n = idx >> 9, k = idx & 511;
    WT[(size_t)n * D_DIM + k] = f32_to_bf16(W[(size_t)k * D_DIM + n]);
}

// ---------------------------------------------------------------------------
// Kernel 2: vectorized f32 -> bf16 conversion of the inputs tensor.
// ---------------------------------------------------------------------------
__global__ void cvt_inputs(const float* __restrict__ in, uint16_t* __restrict__ out) {
    size_t i = ((size_t)blockIdx.x * blockDim.x + threadIdx.x) * 4;
    float4 f = *reinterpret_cast<const float4*>(in + i);
    uint2 p = { pack_bf16(f.x, f.y), pack_bf16(f.z, f.w) };
    *reinterpret_cast<uint2*>(out + i) = p;
}

// ---------------------------------------------------------------------------
// Kernel 3: OUT = act(A @ W + b).  bf16 A; 16x64 per wave (4 acc tiles).
// flags: bit0 = relu, bit1 = f32 output (else bf16).
// ---------------------------------------------------------------------------
__global__ void gemm_tiled(const uint16_t* __restrict__ A, const uint16_t* __restrict__ WT,
                           const float* __restrict__ bias,
                           uint16_t* __restrict__ out16, float* __restrict__ out32,
                           int flags) {
    const int lane = threadIdx.x & 31;
    const int g    = blockIdx.x * 8 + (threadIdx.x >> 5);  // tile-group id
    const int tm   = g >> 3;                               // row tile
    const int tn0  = (g & 7) * 4;                          // first of 4 col tiles

    v8f acc[4] = {{}, {}, {}, {}};
    const uint16_t* a0 = A + (size_t)(tm * 16) * D_DIM;
    for (int kk = 0; kk < KBLK; ++kk) {
        v16bf af = load_fragA(a0 + kk * 32, D_DIM, lane);
#pragma unroll
        for (int u = 0; u < 4; ++u) {
            v16bf bf = load_fragB(WT + (size_t)((tn0 + u) * 16) * D_DIM + kk * 32, D_DIM, lane);
            acc[u] = wmma_bf16(af, bf, acc[u]);
        }
    }
#pragma unroll
    for (int u = 0; u < 4; ++u) {
        const int col = (tn0 + u) * 16 + (lane & 15);
        const float bi = bias[col];
#pragma unroll
        for (int j = 0; j < 8; ++j) {
            int row = tm * 16 + j + ((lane >> 4) << 3);
            float v = acc[u][j] + bi;
            if (flags & 1) v = fmaxf(v, 0.f);
            if (flags & 2) out32[(size_t)row * D_DIM + col] = v;
            else           out16[(size_t)row * D_DIM + col] = f32_to_bf16(v);
        }
    }
}

// ---------------------------------------------------------------------------
// Kernel 4: per-batch fused attention + output projection.
// 256 threads (8 waves) per batch element; 216 KB dynamic LDS.
// ---------------------------------------------------------------------------
#define SQ_OFF   0u                      // 64x512 bf16 Q (reused for ctx)
#define SK_OFF   65536u                  // 64x512 bf16 K
#define SVT_OFF  131072u                 // 512x64 bf16 Vt[d][a]
#define SS_OFF   196608u                 // 64x64  f32  scores
#define SAT_OFF  212992u                 // 64x64  bf16 attn
#define ATT_LDS  221184u

__global__ void attention(const uint16_t* __restrict__ Q, const uint16_t* __restrict__ K,
                          const uint16_t* __restrict__ V, const float* __restrict__ R,
                          const uint16_t* __restrict__ WTo, const float* __restrict__ b_o,
                          float* __restrict__ out) {
    extern __shared__ char smem[];
    uint16_t* sQ   = reinterpret_cast<uint16_t*>(smem + SQ_OFF);
    uint16_t* sK   = reinterpret_cast<uint16_t*>(smem + SK_OFF);
    uint16_t* sVt  = reinterpret_cast<uint16_t*>(smem + SVT_OFF);
    float*    sS   = reinterpret_cast<float*>   (smem + SS_OFF);
    uint16_t* sAt  = reinterpret_cast<uint16_t*>(smem + SAT_OFF);
    uint16_t* sCtx = sQ;

    const int b    = blockIdx.x;
    const int tid  = threadIdx.x;
    const int lane = tid & 31;
    const int wave = tid >> 5;
    const size_t mat = (size_t)b * AGENTS * D_DIM;

    // ---- Phase A: stage Q,K (TDM when available) + Vt transpose ----------
#if __has_builtin(__builtin_amdgcn_tensor_load_to_lds)
    if (wave == 0) {
        tdm_load_bf16_1d(SQ_OFF, Q + mat, AGENTS * D_DIM);
        tdm_load_bf16_1d(SK_OFF, K + mat, AGENTS * D_DIM);
    }
#else
    {
        const uint4* gq = reinterpret_cast<const uint4*>(Q + mat);
        const uint4* gk = reinterpret_cast<const uint4*>(K + mat);
        uint4* sq4 = reinterpret_cast<uint4*>(sQ);
        uint4* sk4 = reinterpret_cast<uint4*>(sK);
        for (int i = tid; i < 4096; i += 256) { sq4[i] = gq[i]; sk4[i] = gk[i]; }
    }
#endif
    {
        const uint32_t* gv = reinterpret_cast<const uint32_t*>(V + mat);
        for (int i = tid; i < 16384; i += 256) {           // V -> Vt[d][a]
            int a = i >> 8, d0 = (i & 255) * 2;
            uint32_t u = gv[a * 256 + (i & 255)];
            sVt[(size_t)d0 * AGENTS + a]       = (uint16_t)(u & 0xFFFFu);
            sVt[(size_t)(d0 + 1) * AGENTS + a] = (uint16_t)(u >> 16);
        }
    }
#if __has_builtin(__builtin_amdgcn_s_wait_tensorcnt)
    __builtin_amdgcn_s_wait_tensorcnt(0);
#endif
    __syncthreads();

    // ---- Phase B: S = Q K^T (waves 0..3: row tile = wave, 4 col tiles) ---
    if (wave < 4) {
        const int tm = wave;
        v8f acc[4] = {{}, {}, {}, {}};
        const uint16_t* a0 = sQ + (size_t)(tm * 16) * D_DIM;
        for (int kk = 0; kk < KBLK; ++kk) {
            v16bf af = load_fragA(a0 + kk * 32, D_DIM, lane);
#pragma unroll
            for (int u = 0; u < 4; ++u)
                acc[u] = wmma_bf16(af, load_fragB(sK + (size_t)(u * 16) * D_DIM + kk * 32,
                                                  D_DIM, lane), acc[u]);
        }
#pragma unroll
        for (int u = 0; u < 4; ++u) {
            int col = u * 16 + (lane & 15);
#pragma unroll
            for (int j = 0; j < 8; ++j)
                sS[(tm * 16 + j + ((lane >> 4) << 3)) * AGENTS + col] = acc[u][j];
        }
    }
    __syncthreads();

    // ---- Phase C: row softmax, emit packed bf16 attn ---------------------
    if (tid < AGENTS) {
        float m = -INFINITY;
        for (int c = 0; c < AGENTS; ++c) m = fmaxf(m, sS[tid * AGENTS + c]);
        float s = 0.f;
        for (int c = 0; c < AGENTS; ++c) {
            float e = expf(sS[tid * AGENTS + c] - m);
            sS[tid * AGENTS + c] = e;
            s += e;
        }
        float inv = 1.f / s;
        for (int c = 0; c < AGENTS; c += 2) {
            uint32_t p = pack_bf16(sS[tid * AGENTS + c] * inv, sS[tid * AGENTS + c + 1] * inv);
            *reinterpret_cast<uint32_t*>(sAt + tid * AGENTS + c) = p;
        }
    }
    __syncthreads();

    // ---- Phase D: ctx = attn V + R -> bf16 in LDS ------------------------
    for (int q = 0; q < 4; ++q) {                          // 32 groups of 4 tiles
        int gg = wave * 4 + q, tm = gg >> 3, tn0 = (gg & 7) * 4;
        v8f acc[4] = {{}, {}, {}, {}};
        for (int kk = 0; kk < 2; ++kk) {
            v16bf af = load_fragA(sAt + (size_t)(tm * 16) * AGENTS + kk * 32, AGENTS, lane);
#pragma unroll
            for (int u = 0; u < 4; ++u)
                acc[u] = wmma_bf16(af, load_fragB(sVt + (size_t)((tn0 + u) * 16) * AGENTS + kk * 32,
                                                  AGENTS, lane), acc[u]);
        }
        const float* r0 = R + mat;
#pragma unroll
        for (int u = 0; u < 4; ++u) {
            int col = (tn0 + u) * 16 + (lane & 15);
#pragma unroll
            for (int j = 0; j < 8; ++j) {
                int row = tm * 16 + j + ((lane >> 4) << 3);
                float v = acc[u][j] + r0[(size_t)row * D_DIM + col];
                sCtx[(size_t)row * D_DIM + col] = f32_to_bf16(v);
            }
        }
    }
    __syncthreads();

    // ---- Phase E: out = relu(ctx W_o + b_o) ------------------------------
    for (int q = 0; q < 4; ++q) {
        int gg = wave * 4 + q, tm = gg >> 3, tn0 = (gg & 7) * 4;
        v8f acc[4] = {{}, {}, {}, {}};
        const uint16_t* a0 = sCtx + (size_t)(tm * 16) * D_DIM;
        for (int kk = 0; kk < KBLK; ++kk) {
            v16bf af = load_fragA(a0 + kk * 32, D_DIM, lane);
#pragma unroll
            for (int u = 0; u < 4; ++u)
                acc[u] = wmma_bf16(af, load_fragB(WTo + (size_t)((tn0 + u) * 16) * D_DIM + kk * 32,
                                                  D_DIM, lane), acc[u]);
        }
#pragma unroll
        for (int u = 0; u < 4; ++u) {
            int col = (tn0 + u) * 16 + (lane & 15);
            const float bi = b_o[col];
#pragma unroll
            for (int j = 0; j < 8; ++j) {
                int row = tm * 16 + j + ((lane >> 4) << 3);
                out[mat + (size_t)row * D_DIM + col] = fmaxf(acc[u][j] + bi, 0.f);
            }
        }
    }
}

// ---------------------------------------------------------------------------
// Host side.  Workspace (~227 MB):
//   6x bf16 weights (3 MB) | INbf16/X/Q/K/V bf16 (5x32 MB) | R f32 (64 MB)
// ---------------------------------------------------------------------------
extern "C" void kernel_launch(void* const* d_in, const int* in_sizes, int n_in,
                              void* d_out, int out_size, void* d_ws, size_t ws_size,
                              hipStream_t stream) {
    const float* inputs = (const float*)d_in[0];
    const float* W_in = (const float*)d_in[1];  const float* b_in = (const float*)d_in[2];
    const float* W_q  = (const float*)d_in[3];  const float* b_q  = (const float*)d_in[4];
    const float* W_k  = (const float*)d_in[5];  const float* b_k  = (const float*)d_in[6];
    const float* W_v  = (const float*)d_in[7];  const float* b_v  = (const float*)d_in[8];
    const float* W_r  = (const float*)d_in[9];  const float* b_r  = (const float*)d_in[10];
    const float* W_o  = (const float*)d_in[11]; const float* b_o  = (const float*)d_in[12];
    float* out = (float*)d_out;

    char* ws = (char*)d_ws;
    const size_t WT_SZ = (size_t)D_DIM * D_DIM * 2;
    uint16_t* WTin = (uint16_t*)(ws + 0 * WT_SZ);
    uint16_t* WTq  = (uint16_t*)(ws + 1 * WT_SZ);
    uint16_t* WTk  = (uint16_t*)(ws + 2 * WT_SZ);
    uint16_t* WTv  = (uint16_t*)(ws + 3 * WT_SZ);
    uint16_t* WTr  = (uint16_t*)(ws + 4 * WT_SZ);
    uint16_t* WTo  = (uint16_t*)(ws + 5 * WT_SZ);
    size_t off = 6 * WT_SZ;
    const size_t MAT16 = (size_t)M_ROWS * D_DIM * 2;       // 32 MB
    uint16_t* INB = (uint16_t*)(ws + off);          off += MAT16;
    uint16_t* X   = (uint16_t*)(ws + off);          off += MAT16;
    uint16_t* Q   = (uint16_t*)(ws + off);          off += MAT16;
    uint16_t* K   = (uint16_t*)(ws + off);          off += MAT16;
    uint16_t* V   = (uint16_t*)(ws + off);          off += MAT16;
    float*    R   = (float*)   (ws + off);          // 64 MB

    // 1) weight transpose+convert and input conversion
    dim3 cb(256), cg(D_DIM * D_DIM / 256);
    wt_convert<<<cg, cb, 0, stream>>>(W_in, WTin);
    wt_convert<<<cg, cb, 0, stream>>>(W_q,  WTq);
    wt_convert<<<cg, cb, 0, stream>>>(W_k,  WTk);
    wt_convert<<<cg, cb, 0, stream>>>(W_v,  WTv);
    wt_convert<<<cg, cb, 0, stream>>>(W_r,  WTr);
    wt_convert<<<cg, cb, 0, stream>>>(W_o,  WTo);
    cvt_inputs<<<dim3((M_ROWS * D_DIM) / (256 * 4)), cb, 0, stream>>>(inputs, INB);

    // 2) X = relu(in W_in + b);  Q,K,V bf16;  R f32
    dim3 gb(256), gg(MT * (NT / 4) / 8);                   // 2048 blocks
    gemm_tiled<<<gg, gb, 0, stream>>>(INB, WTin, b_in, X, nullptr, 1);
    gemm_tiled<<<gg, gb, 0, stream>>>(X, WTq, b_q, Q, nullptr, 0);
    gemm_tiled<<<gg, gb, 0, stream>>>(X, WTk, b_k, K, nullptr, 0);
    gemm_tiled<<<gg, gb, 0, stream>>>(X, WTv, b_v, V, nullptr, 0);
    gemm_tiled<<<gg, gb, 0, stream>>>(X, WTr, b_r, nullptr, R, 2);

    // 3) fused attention + output projection (216 KB dynamic LDS)
    hipFuncSetAttribute((const void*)attention,
                        hipFuncAttributeMaxDynamicSharedMemorySize, ATT_LDS);
    attention<<<dim3(BATCH), dim3(256), ATT_LDS, stream>>>(Q, K, V, R, WTo, b_o, out);
}
